// ProposedModel_82325933129729
// MI455X (gfx1250) — matrix operand order repaired and checked by reading, not compiled, not run on hardware
//
#include <hip/hip_runtime.h>
#include <stdint.h>

// ---------------- model constants ----------------
#define NL   6
#define DMODEL 768
#define NH   8
#define DQ   96
#define DFF  3072
#define ODIM 80
#define NRELV 201
#define BSZ  8
#define TS   512
#define MROWS (BSZ*TS)   // 4096

typedef __bf16 bf16_t;
typedef __attribute__((ext_vector_type(8)))  bf16_t v8bf;
typedef __attribute__((ext_vector_type(16))) bf16_t v16bf;
typedef __attribute__((ext_vector_type(8)))  float  v8f;

union BF16x16 { v16bf v; v8bf h[2]; };

__device__ __forceinline__ bf16_t f2bf(float f) {
  union { float f; uint32_t u; } in; in.f = f;
  uint32_t r = (in.u + 0x7FFFu + ((in.u >> 16) & 1u)) >> 16;
  union { uint16_t u; bf16_t b; } out; out.u = (uint16_t)r;
  return out.b;
}

// CDNA5 async memory->LDS copy (ASYNCcnt-tracked).  The LDS destination is a
// per-lane VGPR holding the LDS byte address; generic pointers to __shared__
// carry the LDS offset in their low 32 bits (SHARED aperture, ISA 10.2).
__device__ __forceinline__ void async_b128(uint64_t gaddr, uint32_t loff) {
  asm volatile("global_load_async_to_lds_b128 %0, %1, off"
               :: "v"(loff), "v"(gaddr) : "memory");
}

// ---------------- generic bf16 WMMA GEMM ----------------
// C[M,N] = scale * (A[M,K] x B) + bias  (optional relu; f32 or bf16 out)
// flags&1 : B stored as [N][K] row-major, else B stored as [K][N] row-major.
// Batched over blockIdx.z with element strides sA/sB/sC.
// Requirements: M % 128 == 0, K % 32 == 0, lda/ldb multiples of 8. N guarded.
#define FLAG_TRB  1
#define FLAG_RELU 2
#define FLAG_OBF  4

__global__ __launch_bounds__(256) void k_gemm(
    const bf16_t* __restrict__ A, const bf16_t* __restrict__ B,
    const float* __restrict__ bias,
    float* __restrict__ Cf, bf16_t* __restrict__ Cb,
    int N, int K, int lda, int ldb, int ldc,
    long long sA, long long sB, long long sC,
    float scale, int flags)
{
  // double-buffered tiles: 128 rows x 32 (+8 pad) bf16
  __shared__ alignas(16) bf16_t As[2][128 * 40];
  __shared__ alignas(16) bf16_t Bs[2][128 * 40];
  __shared__ alignas(16) char   trash[32 * 16];   // sink for masked async lanes

  const int tid  = threadIdx.x;
  const int lane = tid & 31;
  const int wave = tid >> 5;
  const int wm   = wave >> 1;       // 0..3  (32-row slice)
  const int wn   = wave & 1;        // 0..1  (64-col slice)
  const int nl   = lane & 15;
  const int kh   = lane >> 4;       // lane half
  const int bm   = blockIdx.y * 128;
  const int bn   = blockIdx.x * 128;

  A += (long long)blockIdx.z * sA;
  B += (long long)blockIdx.z * sB;
  const long long coff = (long long)blockIdx.z * sC;

  auto fillA = [&](int buf, int k0) {
    #pragma unroll
    for (int p = 0; p < 2; ++p) {
      int t = tid + p * 256;
      int row = t >> 2, seg = (t & 3) << 3;
      uint64_t g = (uint64_t)(uintptr_t)(A + (size_t)(bm + row) * lda + (k0 + seg));
      uint32_t l = (uint32_t)(uintptr_t)&As[buf][row * 40 + seg];
      async_b128(g, l);
    }
  };
  auto fillB_trb = [&](int buf, int k0) {   // B is [N][K]: direct K-contiguous rows
    #pragma unroll
    for (int p = 0; p < 2; ++p) {
      int t = tid + p * 256;
      int row = t >> 2, seg = (t & 3) << 3;
      uint64_t g; uint32_t l;
      if (bn + row < N) {
        g = (uint64_t)(uintptr_t)(B + (size_t)(bn + row) * ldb + (k0 + seg));
        l = (uint32_t)(uintptr_t)&Bs[buf][row * 40 + seg];
      } else {                               // keep async count uniform
        g = (uint64_t)(uintptr_t)B;
        l = (uint32_t)(uintptr_t)&trash[(tid & 31) * 16];
      }
      async_b128(g, l);
    }
  };
  auto fillB_kn = [&](int buf, int k0) {    // B is [K][N]: transpose via LDS scatter
    #pragma unroll
    for (int p = 0; p < 2; ++p) {
      int t  = tid + p * 256;
      int kr = t >> 4, ns = (t & 15) << 3;
      if (bn + ns + 7 < N) {
        v8bf vb = *(const v8bf*)(B + (size_t)(k0 + kr) * ldb + (bn + ns));
        #pragma unroll
        for (int e = 0; e < 8; ++e) Bs[buf][(ns + e) * 40 + kr] = vb[e];
      } else {
        #pragma unroll
        for (int e = 0; e < 8; ++e) {
          bf16_t x = f2bf(0.0f);
          if (bn + ns + e < N) x = B[(size_t)(k0 + kr) * ldb + (bn + ns + e)];
          Bs[buf][(ns + e) * 40 + kr] = x;
        }
      }
    }
  };

  // zero OOB B rows once (never rewritten by the async path: those go to trash)
  if ((flags & FLAG_TRB) && (N - bn < 128)) {
    bf16_t z = f2bf(0.0f);
    for (int n = tid; n < 128; n += 256) {
      if (bn + n >= N) {
        #pragma unroll
        for (int e = 0; e < 32; ++e) { Bs[0][n * 40 + e] = z; Bs[1][n * 40 + e] = z; }
      }
    }
  }

  v8f acc[2][4];
  #pragma unroll
  for (int i = 0; i < 2; ++i)
    #pragma unroll
    for (int j = 0; j < 4; ++j)
      #pragma unroll
      for (int e = 0; e < 8; ++e) acc[i][j][e] = 0.0f;

  const int steps = K >> 5;
  fillA(0, 0);
  if (flags & FLAG_TRB) fillB_trb(0, 0); else fillB_kn(0, 0);
  asm volatile("s_wait_asynccnt 0" ::: "memory");
  __syncthreads();

  for (int i = 0; i < steps; ++i) {
    const int cur = i & 1;
    if (i + 1 < steps) {
      int kn = (i + 1) << 5;
      fillA(cur ^ 1, kn);
      if (flags & FLAG_TRB) {
        fillB_trb(cur ^ 1, kn);
        // 4 async ops just issued for next tile; in-order completion means
        // asynccnt<=4 guarantees this tile's copies have landed.
        asm volatile("s_wait_asynccnt 4" ::: "memory");
      } else {
        fillB_kn(cur ^ 1, kn);
        asm volatile("s_wait_asynccnt 2" ::: "memory");
      }
    } else {
      asm volatile("s_wait_asynccnt 0" ::: "memory");
    }
    __syncthreads();

    // ---- fragments per CDNA5 VGPR layouts
    BF16x16 af[2], bfr[4];
    #pragma unroll
    for (int ii = 0; ii < 2; ++ii) {
      int m = wm * 32 + ii * 16 + nl;
      af[ii].h[0] = *(const v8bf*)(&As[cur][m * 40 + kh * 8]);       // K 0..7 / 8..15
      af[ii].h[1] = *(const v8bf*)(&As[cur][m * 40 + 16 + kh * 8]);  // K 16..23 / 24..31
    }
    #pragma unroll
    for (int j = 0; j < 4; ++j) {
      int n = wn * 64 + j * 16 + nl;
      bfr[j].h[0] = *(const v8bf*)(&Bs[cur][n * 40 + kh * 8]);       // K 0..7 / 8..15
      bfr[j].h[1] = *(const v8bf*)(&Bs[cur][n * 40 + 16 + kh * 8]);  // K 16..23 / 24..31
    }
    #pragma unroll
    for (int ii = 0; ii < 2; ++ii)
      #pragma unroll
      for (int j = 0; j < 4; ++j)
        acc[ii][j] = __builtin_amdgcn_wmma_f32_16x16x32_bf16(
            false, af[ii].v, false, bfr[j].v, (short)0, acc[ii][j], false, false);
    __syncthreads();   // reads done before this buffer is refilled
  }

  // ---- epilogue
  #pragma unroll
  for (int i = 0; i < 2; ++i) {
    #pragma unroll
    for (int j = 0; j < 4; ++j) {
      int col = bn + wn * 64 + j * 16 + nl;
      if (col < N) {
        float bv = bias ? bias[col] : 0.0f;
        #pragma unroll
        for (int r = 0; r < 8; ++r) {
          int row = bm + wm * 32 + i * 16 + kh * 8 + r;   // lane half -> M+8
          float v = acc[i][j][r] * scale + bv;
          if (flags & FLAG_RELU) v = fmaxf(v, 0.0f);
          long long idx = coff + (long long)row * ldc + col;
          if (flags & FLAG_OBF) Cb[idx] = f2bf(v);
          else                  Cf[idx] = v;
        }
      }
    }
  }
}

// ---------------- block reductions ----------------
__device__ __forceinline__ float blk_sum(float v, float* red) {
  int tid = threadIdx.x;
  red[tid] = v; __syncthreads();
  #pragma unroll
  for (int s = 128; s > 0; s >>= 1) {
    if (tid < s) red[tid] += red[tid + s];
    __syncthreads();
  }
  float r = red[0]; __syncthreads();
  return r;
}
__device__ __forceinline__ float blk_max(float v, float* red) {
  int tid = threadIdx.x;
  red[tid] = v; __syncthreads();
  #pragma unroll
  for (int s = 128; s > 0; s >>= 1) {
    if (tid < s) red[tid] = fmaxf(red[tid], red[tid + s]);
    __syncthreads();
  }
  float r = red[0]; __syncthreads();
  return r;
}

// ---------------- softmax with relative-position gather ----------------
// S: [H,512,512] f32 ; relQ: [H,512,201] f32 ; P: [H,512,512] bf16
// rel[q,k] = relQ[q, k-q+99] if 0 <= k-q+99 < 199 else -1e8
__global__ __launch_bounds__(256) void k_softmax(
    const float* __restrict__ S, const float* __restrict__ relQ,
    bf16_t* __restrict__ P)
{
  __shared__ float red[256];
  int q = blockIdx.x, h = blockIdx.y, tid = threadIdx.x;
  const float* srow = S    + ((size_t)h * TS + q) * TS;
  const float* rrow = relQ + ((size_t)h * TS + q) * NRELV;
  float v[2];
  #pragma unroll
  for (int j = 0; j < 2; ++j) {
    int c = tid + 256 * j;
    int d = c - q + 99;
    float rel = ((unsigned)d < 199u) ? rrow[d] : -1.0e8f;
    v[j] = srow[c] + rel;
  }
  float m  = blk_max(fmaxf(v[0], v[1]), red);
  float e0 = __expf(v[0] - m), e1 = __expf(v[1] - m);
  float sm = blk_sum(e0 + e1, red);
  float inv = 1.0f / sm;
  bf16_t* prow = P + ((size_t)h * TS + q) * TS;
  prow[tid]       = f2bf(e0 * inv);
  prow[tid + 256] = f2bf(e1 * inv);
}

// ---------------- residual + LayerNorm (row = 768) ----------------
__global__ __launch_bounds__(256) void k_add_ln(
    const float* __restrict__ src, const float* __restrict__ delta,
    const float* __restrict__ g, const float* __restrict__ bta,
    float* __restrict__ out, bf16_t* __restrict__ outb)
{
  __shared__ float red[256];
  int row = blockIdx.x, tid = threadIdx.x;
  float x[3]; float s = 0.0f, s2 = 0.0f;
  #pragma unroll
  for (int j = 0; j < 3; ++j) {
    int c = tid + 256 * j;
    float v = src[(size_t)row * DMODEL + c] + delta[(size_t)row * DMODEL + c];
    x[j] = v; s += v; s2 += v * v;
  }
  float st  = blk_sum(s,  red);
  float s2t = blk_sum(s2, red);
  float mu  = st * (1.0f / DMODEL);
  float var = s2t * (1.0f / DMODEL) - mu * mu;
  float rs  = rsqrtf(var + 1e-5f);
  #pragma unroll
  for (int j = 0; j < 3; ++j) {
    int c = tid + 256 * j;
    float y = (x[j] - mu) * rs * g[c] + bta[c];
    out [(size_t)row * DMODEL + c] = y;
    outb[(size_t)row * DMODEL + c] = f2bf(y);
  }
}

// ---------------- conversions ----------------
__global__ void k_cvt(const float* __restrict__ in, bf16_t* __restrict__ out, int n) {
  int i = blockIdx.x * 256 + threadIdx.x;
  if (i < n) out[i] = f2bf(in[i]);
}
// wq/wk/wv (H, F, A) -> [N=H*A][K=F]
__global__ void k_cvt_qkvT(const float* __restrict__ in, bf16_t* __restrict__ out) {
  int i = blockIdx.x * 256 + threadIdx.x;
  if (i < DMODEL * DMODEL) {
    int nn = i / DMODEL, f = i % DMODEL;
    int h = nn / DQ, a = nn % DQ;
    out[i] = f2bf(in[((size_t)h * DMODEL + f) * DQ + a]);
  }
}

// ---------------- host helpers ----------------
static inline void gemmL(const bf16_t* A, const bf16_t* B, const float* bias,
                         void* C, int M, int N, int K, int lda, int ldb, int ldc,
                         long long sA, long long sB, long long sC, int batch,
                         float scale, int flags, hipStream_t s)
{
  dim3 g((N + 127) / 128, M / 128, batch), b(256, 1, 1);
  float*  cf = (flags & FLAG_OBF) ? nullptr : (float*)C;
  bf16_t* cb = (flags & FLAG_OBF) ? (bf16_t*)C : nullptr;
  k_gemm<<<g, b, 0, s>>>(A, B, bias, cf, cb, N, K, lda, ldb, ldc, sA, sB, sC, scale, flags);
}

extern "C" void kernel_launch(void* const* d_in, const int* in_sizes, int n_in,
                              void* d_out, int out_size, void* d_ws, size_t ws_size,
                              hipStream_t stream)
{
  (void)in_sizes; (void)n_in; (void)out_size; (void)ws_size;
  const float* x     = (const float*)d_in[0];
  const float* wq    = (const float*)d_in[1];
  const float* wk    = (const float*)d_in[2];
  const float* wv    = (const float*)d_in[3];
  const float* wo    = (const float*)d_in[4];
  const float* remb  = (const float*)d_in[5];
  const float* ff1w  = (const float*)d_in[6];
  const float* ff1bi = (const float*)d_in[7];
  const float* ff2w  = (const float*)d_in[8];
  const float* ff2bi = (const float*)d_in[9];
  const float* ln1g  = (const float*)d_in[10];
  const float* ln1bb = (const float*)d_in[11];
  const float* ln2g  = (const float*)d_in[12];
  const float* ln2bb = (const float*)d_in[13];
  const float* woutw = (const float*)d_in[14];
  const float* boutv = (const float*)d_in[15];

  char* w = (char*)d_ws;
  size_t off = 0;
  auto alloc = [&](size_t bytes) -> void* {
    off = (off + 255) & ~(size_t)255;
    void* p = w + off; off += bytes; return p;
  };
  float*  srcf  = (float*)  alloc((size_t)MROWS * DMODEL * 4);
  bf16_t* xb    = (bf16_t*) alloc((size_t)MROWS * DMODEL * 2);
  bf16_t* qb    = (bf16_t*) alloc((size_t)MROWS * DMODEL * 2);
  bf16_t* kb    = (bf16_t*) alloc((size_t)MROWS * DMODEL * 2);
  bf16_t* vb    = (bf16_t*) alloc((size_t)MROWS * DMODEL * 2);
  bf16_t* ob    = (bf16_t*) alloc((size_t)MROWS * DMODEL * 2);
  float*  attn  = (float*)  alloc((size_t)MROWS * DMODEL * 4);
  bf16_t* hb    = (bf16_t*) alloc((size_t)MROWS * DFF * 2);
  float*  ffb   = (float*)  alloc((size_t)MROWS * DMODEL * 4);
  bf16_t* wqt   = (bf16_t*) alloc((size_t)DMODEL * DMODEL * 2);
  bf16_t* wkt   = (bf16_t*) alloc((size_t)DMODEL * DMODEL * 2);
  bf16_t* wvt   = (bf16_t*) alloc((size_t)DMODEL * DMODEL * 2);
  bf16_t* wob   = (bf16_t*) alloc((size_t)DMODEL * DMODEL * 2);
  bf16_t* embb  = (bf16_t*) alloc((size_t)NH * NRELV * DQ * 2);
  bf16_t* f1b   = (bf16_t*) alloc((size_t)DFF * DMODEL * 2);
  bf16_t* f2bw  = (bf16_t*) alloc((size_t)DMODEL * DFF * 2);
  bf16_t* woutb = (bf16_t*) alloc((size_t)ODIM * DMODEL * 2);
  float*  relQ  = (float*)  alloc((size_t)NH * TS * NRELV * 4);
  float*  logit = (float*)  alloc((size_t)NH * TS * TS * 4);
  bf16_t* Pbuf  = (bf16_t*) alloc((size_t)NH * TS * TS * 2);

  const float qscale = 0.10206207261596575f;   // 1/sqrt(96)

  hipMemcpyAsync(srcf, x, (size_t)MROWS * DMODEL * 4, hipMemcpyDeviceToDevice, stream);
  k_cvt<<<(MROWS * DMODEL + 255) / 256, 256, 0, stream>>>(x, xb, MROWS * DMODEL);
  k_cvt<<<(ODIM * DMODEL + 255) / 256, 256, 0, stream>>>(woutw, woutb, ODIM * DMODEL);

  for (int l = 0; l < NL; ++l) {
    const float* wq_l  = wq   + (size_t)l * NH * DMODEL * DQ;
    const float* wk_l  = wk   + (size_t)l * NH * DMODEL * DQ;
    const float* wv_l  = wv   + (size_t)l * NH * DMODEL * DQ;
    const float* wo_l  = wo   + (size_t)l * NH * DQ * DMODEL;
    const float* emb_l = remb + (size_t)l * NH * NRELV * DQ;
    const float* f1_l  = ff1w + (size_t)l * DFF * DMODEL;
    const float* f2_l  = ff2w + (size_t)l * DMODEL * DFF;

    int g768 = (DMODEL * DMODEL + 255) / 256;
    k_cvt_qkvT<<<g768, 256, 0, stream>>>(wq_l, wqt);
    k_cvt_qkvT<<<g768, 256, 0, stream>>>(wk_l, wkt);
    k_cvt_qkvT<<<g768, 256, 0, stream>>>(wv_l, wvt);
    k_cvt<<<g768, 256, 0, stream>>>(wo_l, wob, DMODEL * DMODEL);
    k_cvt<<<(NH * NRELV * DQ + 255) / 256, 256, 0, stream>>>(emb_l, embb, NH * NRELV * DQ);
    k_cvt<<<(DFF * DMODEL + 255) / 256, 256, 0, stream>>>(f1_l, f1b, DFF * DMODEL);
    k_cvt<<<(DMODEL * DFF + 255) / 256, 256, 0, stream>>>(f2_l, f2bw, DMODEL * DFF);

    // Q,K,V projections: [4096,768] x [768,768]  (q pre-scaled)
    gemmL(xb, wqt, nullptr, qb, MROWS, DMODEL, DMODEL, DMODEL, DMODEL, DMODEL,
          0, 0, 0, 1, qscale, FLAG_TRB | FLAG_OBF, stream);
    gemmL(xb, wkt, nullptr, kb, MROWS, DMODEL, DMODEL, DMODEL, DMODEL, DMODEL,
          0, 0, 0, 1, 1.0f, FLAG_TRB | FLAG_OBF, stream);
    gemmL(xb, wvt, nullptr, vb, MROWS, DMODEL, DMODEL, DMODEL, DMODEL, DMODEL,
          0, 0, 0, 1, 1.0f, FLAG_TRB | FLAG_OBF, stream);

    for (int b = 0; b < BSZ; ++b) {
      const bf16_t* qb_b = qb + (size_t)b * TS * DMODEL;
      const bf16_t* kb_b = kb + (size_t)b * TS * DMODEL;
      const bf16_t* vb_b = vb + (size_t)b * TS * DMODEL;
      bf16_t*       ob_b = ob + (size_t)b * TS * DMODEL;
      // relQ[h,t,j] = q[h,t,:] . emb[h,j,:]   ([512,96]x[96,201], batched over h)
      gemmL(qb_b, embb, nullptr, relQ, TS, NRELV, DQ, DMODEL, DQ, NRELV,
            96, (long long)NRELV * DQ, (long long)TS * NRELV, NH,
            1.0f, FLAG_TRB, stream);
      // S[h,q,k] = q . k              ([512,96]x[96,512], batched over h)
      gemmL(qb_b, kb_b, nullptr, logit, TS, TS, DQ, DMODEL, DMODEL, TS,
            96, 96, (long long)TS * TS, NH, 1.0f, FLAG_TRB, stream);
      // softmax(S + rel) -> P (bf16)
      k_softmax<<<dim3(TS, NH), 256, 0, stream>>>(logit, relQ, Pbuf);
      // O = P x V                     ([512,512]x[512,96], batched over h)
      gemmL(Pbuf, vb_b, nullptr, ob_b, TS, DQ, TS, TS, DMODEL, DMODEL,
            (long long)TS * TS, 96, 96, NH, 1.0f, FLAG_OBF, stream);
    }

    // attn = O x Wo   ([4096,768]x[768,768], Wo is [K][N] natural)
    gemmL(ob, wob, nullptr, attn, MROWS, DMODEL, DMODEL, DMODEL, DMODEL, DMODEL,
          0, 0, 0, 1, 1.0f, 0, stream);
    k_add_ln<<<MROWS, 256, 0, stream>>>(srcf, attn, ln1g + (size_t)l * DMODEL,
                                        ln1bb + (size_t)l * DMODEL, srcf, xb);

    // FFN
    gemmL(xb, f1b, ff1bi + (size_t)l * DFF, hb, MROWS, DFF, DMODEL,
          DMODEL, DMODEL, DFF, 0, 0, 0, 1, 1.0f,
          FLAG_TRB | FLAG_RELU | FLAG_OBF, stream);
    gemmL(hb, f2bw, ff2bi + (size_t)l * DMODEL, ffb, MROWS, DMODEL, DFF,
          DFF, DFF, DMODEL, 0, 0, 0, 1, 1.0f, FLAG_TRB, stream);
    k_add_ln<<<MROWS, 256, 0, stream>>>(srcf, ffb, ln2g + (size_t)l * DMODEL,
                                        ln2bb + (size_t)l * DMODEL, srcf, xb);
  }

  // out = src x wout^T + bout   ([4096,768]x[768,80])
  gemmL(xb, woutb, boutv, d_out, MROWS, ODIM, DMODEL, DMODEL, DMODEL, ODIM,
        0, 0, 0, 1, 1.0f, FLAG_TRB, stream);
}
